// MambaAdapter_73418170958497
// MI455X (gfx1250) — matrix operand order repaired and checked by reading, not compiled, not run on hardware
//
#include <hip/hip_runtime.h>
#include <hip/hip_bf16.h>
#include <math.h>

// ---------------- problem constants ----------------
#define B_SZ  8
#define L_SZ  2048
#define D_SZ  1024
#define BD    512
#define DIN   1024
#define NST   16
#define KC    4
#define DTR   32
#define NTOK  (B_SZ * L_SZ)   // 16384 tokens

// ---------------- WMMA / TDM types (gfx1250, wave32) ----------------
typedef __attribute__((ext_vector_type(16))) __bf16 v16bf;
typedef __attribute__((ext_vector_type(8)))  __bf16 v8bf;
typedef __attribute__((ext_vector_type(8)))  float  v8f;
typedef __attribute__((ext_vector_type(4)))  unsigned int v4u;
typedef __attribute__((ext_vector_type(8)))  int v8i;
typedef __attribute__((ext_vector_type(4)))  int v4i;

// ---------------- small helper kernels ----------------
__global__ __launch_bounds__(256) void cvt_bf16_kernel(
    const float* __restrict__ in, __bf16* __restrict__ out, int n) {
  int i = blockIdx.x * 256 + threadIdx.x;
  if (i < n) out[i] = (__bf16)in[i];
}

__global__ __launch_bounds__(256) void aneg_kernel(
    const float* __restrict__ alog, float* __restrict__ aneg, int n) {
  int i = blockIdx.x * 256 + threadIdx.x;
  if (i < n) aneg[i] = -__expf(alog[i]);
}

// ---------------- WMMA bf16 GEMM ----------------
// C[M,N] = A[M,K] * W[K,N](bf16 row-major) (+epilogue)
// Block: 256 thr = 8 waves; tile 128x128; K chunked by 32.
// Wave grid 4(M)x2(N); each wave: 2 Mtiles x 4 Ntiles of 16x16.
#define BM  128
#define BN  128
#define BK  32
#define PAD 40   // LDS row pitch (bf16 elems) == TDM pad: 16 DW data + 4 DW pad

// A fragment (16x32 bf16): lane l -> row (l&15); hi=l>>4:
//   elems 0..7  = K[hi*8 .. +7], elems 8..15 = K[16+hi*8 .. +7]
static __device__ inline v16bf ld_fragA(const __bf16* rowp, int hi) {
  v8bf lo = *(const v8bf*)(rowp + hi * 8);
  v8bf hh = *(const v8bf*)(rowp + 16 + hi * 8);
  v16bf r;
#pragma unroll
  for (int j = 0; j < 8; ++j) { r[j] = lo[j]; r[j + 8] = hh[j]; }
  return r;
}

// B fragment (32x16 bf16): lane l -> col (l&15); hi=l>>4:
//   elems 0..7  = K[hi*8 .. +7], elems 8..15 = K[16+hi*8 .. +7]
static __device__ inline v16bf ld_fragB(const __bf16* rowp, int hi) {
  v8bf lo = *(const v8bf*)(rowp + hi * 8);
  v8bf hh = *(const v8bf*)(rowp + 16 + hi * 8);
  v16bf r;
#pragma unroll
  for (int j = 0; j < 8; ++j) { r[j] = lo[j]; r[j + 8] = hh[j]; }
  return r;
}

// ---- A staging, fp32 source: global -> cvt -> LDS (per-thread) ----
static __device__ inline void stage_A(const float* A, int lda, int M, int K,
                                      int bm0, int kc, int tid,
                                      __bf16 (*sA)[PAD]) {
  const int r = tid >> 1, seg = (tid & 1) * 16;
  const float* ap = A + (size_t)(bm0 + r) * lda + kc + seg;
#pragma unroll
  for (int j = 0; j < 16; j += 4) {
    float4 v = *(const float4*)(ap + j);
    sA[r][seg + j + 0] = (__bf16)v.x;
    sA[r][seg + j + 1] = (__bf16)v.y;
    sA[r][seg + j + 2] = (__bf16)v.z;
    sA[r][seg + j + 3] = (__bf16)v.w;
  }
}

// ---- A staging, bf16 source: Tensor Data Mover (async DMA -> LDS) ----
// 2D D#: data_size=2B, tile 32(K) x 128(M), tensor stride0 = lda elems,
// LDS pad: every 16 DWORDs insert 4 DWORDs -> row pitch 80B == PAD*2.
static __device__ inline void stage_A(const __bf16* A, int lda, int M, int K,
                                      int bm0, int kc, int tid,
                                      __bf16 (*sA)[PAD]) {
  if (tid < 32) {  // wave 0 issues the DMA; EXEC-independent, wave-level op
    unsigned lds_off = (unsigned)(size_t)(&sA[0][0]);  // low 32b = LDS offset
    unsigned long long ga =
        (unsigned long long)(size_t)(A + (size_t)bm0 * lda + kc);
    v4u g0;
    g0[0] = 1u;                                   // count=1, no gather
    g0[1] = lds_off;                              // lds_addr
    g0[2] = (unsigned)(ga & 0xFFFFFFFFu);         // global_addr[31:0]
    g0[3] = (unsigned)((ga >> 32) & 0x01FFFFFFu)  // global_addr[56:32]
            | (2u << 30);                         // type=2 ("image")
    const unsigned td0 = (unsigned)K, td1 = (unsigned)M;
    const unsigned long long s0 = (unsigned long long)lda;
    v8i g1;
    g1[0] = (int)((1u << 16)      // data_size = 2B
                  | (1u << 20)    // pad_enable
                  | (3u << 22)    // pad_interval: 16 DWORDs
                  | (3u << 25));  // pad_amount: 4 DWORDs
    g1[1] = (int)((td0 & 0xFFFFu) << 16);                       // tensor_dim0 lo
    g1[2] = (int)(((td0 >> 16) & 0xFFFFu) | ((td1 & 0xFFFFu) << 16));
    g1[3] = (int)(((td1 >> 16) & 0xFFFFu) | (32u << 16));       // tile_dim0=32
    g1[4] = (int)128;                                           // tile_dim1=128
    g1[5] = (int)(unsigned)(s0 & 0xFFFFFFFFu);                  // dim0_stride
    g1[6] = (int)(unsigned)((s0 >> 32) & 0xFFFFu);
    g1[7] = 0;
    v4i gz4 = {0, 0, 0, 0};
    v8i gz8 = {0, 0, 0, 0, 0, 0, 0, 0};
    __builtin_amdgcn_tensor_load_to_lds(g0, g1, gz4, gz4, gz8, 0);
    __builtin_amdgcn_s_wait_tensorcnt(0);
  }
}

// EP: 0 = none, 1 = +bias, 2 = softplus(x + bias)
// AT: float (cvt staging) or __bf16 (TDM staging); CT: float or __bf16 output
template <int EP, typename AT, typename CT>
__global__ __launch_bounds__(256) void gemm_bf16_kernel(
    const AT* __restrict__ A, const __bf16* __restrict__ Wb,
    const float* __restrict__ bias, CT* __restrict__ C,
    int M, int N, int K, int lda, int ldc) {
  __shared__ __bf16 sA[BM][PAD];
  __shared__ __bf16 sB[BN][PAD];

  const int tid  = threadIdx.x;
  const int bm0  = blockIdx.y * BM;
  const int bn0  = blockIdx.x * BN;
  const int wid  = tid >> 5, lane = tid & 31;
  const int wm   = wid & 3, wn = wid >> 2;   // 4 waves over M, 2 over N
  const int lrow = lane & 15, hi = lane >> 4;

  const v8f vzero = {0.f, 0.f, 0.f, 0.f, 0.f, 0.f, 0.f, 0.f};
  v8f acc[2][4];
#pragma unroll
  for (int a = 0; a < 2; ++a)
#pragma unroll
    for (int b = 0; b < 4; ++b) acc[a][b] = vzero;

  for (int kc = 0; kc < K; kc += BK) {
    // --- stage B: 32x128 bf16, transposed into sB[n][k], zero-pad OOB N ---
    {
      const int k = tid >> 3, seg = (tid & 7) * 16;
      const __bf16* bp = Wb + (size_t)(kc + k) * N + bn0 + seg;
      if (bn0 + seg + 16 <= N) {
#pragma unroll
        for (int j = 0; j < 16; ++j) sB[seg + j][k] = bp[j];
      } else {
#pragma unroll
        for (int j = 0; j < 16; ++j) {
          int gn = bn0 + seg + j;
          sB[seg + j][k] = (gn < N) ? bp[j] : (__bf16)0.f;
        }
      }
    }
    // --- stage A (overload: per-thread cvt for fp32, TDM DMA for bf16) ---
    stage_A(A, lda, M, K, bm0, kc, tid, sA);
    __syncthreads();

    v16bf afr[2], bfr[4];
#pragma unroll
    for (int tm = 0; tm < 2; ++tm)
      afr[tm] = ld_fragA(&sA[wm * 32 + tm * 16 + lrow][0], hi);
#pragma unroll
    for (int tn = 0; tn < 4; ++tn)
      bfr[tn] = ld_fragB(&sB[wn * 64 + tn * 16 + lrow][0], hi);

#pragma unroll
    for (int tm = 0; tm < 2; ++tm)
#pragma unroll
      for (int tn = 0; tn < 4; ++tn)
        acc[tm][tn] = __builtin_amdgcn_wmma_f32_16x16x32_bf16(
            false, afr[tm], false, bfr[tn], (short)0, acc[tm][tn], false, false);
    __syncthreads();
  }

  // --- epilogue: lanes 0-15 -> N=lane, M=v; lanes 16-31 -> N=lane-16, M=v+8
#pragma unroll
  for (int tm = 0; tm < 2; ++tm) {
    const int r0 = bm0 + wm * 32 + tm * 16 + hi * 8;
#pragma unroll
    for (int tn = 0; tn < 4; ++tn) {
      const int col = bn0 + wn * 64 + tn * 16 + lrow;
      if (col >= N) continue;
      const float bv = (EP != 0) ? bias[col] : 0.f;
      CT* cp = C + (size_t)r0 * ldc + col;
#pragma unroll
      for (int v = 0; v < 8; ++v) {
        float x = acc[tm][tn][v];
        if (EP >= 1) x += bv;
        if (EP == 2) x = (x > 20.f) ? x : log1pf(__expf(x));
        cp[(size_t)v * ldc] = (CT)x;
      }
    }
  }
}

// ---------------- causal depthwise conv (K=4) + SiLU ----------------
// xz [NTOK, 2*DIN]; writes xc (fp32, for scan/combine) and xcb (bf16, for GEMM)
__global__ __launch_bounds__(256) void conv_silu_kernel(
    const float* __restrict__ xz, const float* __restrict__ cw,
    const float* __restrict__ cb, float* __restrict__ xc,
    __bf16* __restrict__ xcb) {
  size_t i = (size_t)blockIdx.x * 256 + threadIdx.x;
  if (i >= (size_t)NTOK * DIN) return;
  const int d = (int)(i % DIN);
  const size_t row = i / DIN;          // b*L + t
  const int t = (int)(row % L_SZ);
  float acc = cb[d];
#pragma unroll
  for (int k = 0; k < KC; ++k) {
    const int back = (KC - 1) - k;     // taps t-3..t
    if (t >= back)
      acc = fmaf(xz[(row - (size_t)back) * (2 * DIN) + d], cw[d * KC + k], acc);
  }
  acc = acc / (1.f + __expf(-acc));    // SiLU
  xc[i] = acc;
  xcb[i] = (__bf16)acc;
}

// ---------------- selective scan: lane-per-channel, 16 states in regs ----------------
__global__ __launch_bounds__(256) void scan_kernel(
    const float* __restrict__ dt, const float* __restrict__ xc,
    const float* __restrict__ xdbl, const float* __restrict__ aneg,
    float* __restrict__ y) {
  const int b = blockIdx.y;
  const int d = blockIdx.x * 256 + threadIdx.x;
  float Areg[NST], h[NST];
#pragma unroll
  for (int n = 0; n < NST; ++n) { Areg[n] = aneg[d * NST + n]; h[n] = 0.f; }
  size_t row = (size_t)b * L_SZ;
  for (int t = 0; t < L_SZ; ++t, ++row) {
    const float dtv = dt[row * DIN + d];
    const float xv  = xc[row * DIN + d];
    const float dtx = dtv * xv;
    const float* bc = xdbl + row * 64;   // [0:32)=dt_in, [32:48)=B, [48:64)=C
    float yv = 0.f;
#pragma unroll
    for (int n = 0; n < NST; ++n) {
      h[n] = fmaf(h[n], __expf(dtv * Areg[n]), dtx * bc[DTR + n]);
      yv = fmaf(h[n], bc[DTR + NST + n], yv);
    }
    y[row * DIN + d] = yv;
  }
}

// ---------------- ycb = bf16[(y + D_skip*xc) * silu(z)] ----------------
__global__ __launch_bounds__(256) void combine_kernel(
    const float* __restrict__ y, const float* __restrict__ xc,
    const float* __restrict__ xz, const float* __restrict__ dskip,
    __bf16* __restrict__ ycb) {
  size_t i = (size_t)blockIdx.x * 256 + threadIdx.x;
  if (i >= (size_t)NTOK * DIN) return;
  const int d = (int)(i % DIN);
  const size_t row = i / DIN;
  const float z = xz[row * (2 * DIN) + DIN + d];
  const float sz = z / (1.f + __expf(-z));
  ycb[i] = (__bf16)((y[i] + dskip[d] * xc[i]) * sz);
}

// ---------------- LayerNorm(512): wave32 per row, fp32 in -> bf16 out ----------------
__global__ __launch_bounds__(256) void layernorm_kernel(
    const float* __restrict__ m, const float* __restrict__ g,
    const float* __restrict__ bb, __bf16* __restrict__ mout) {
  const int wid = threadIdx.x >> 5, lane = threadIdx.x & 31;
  const size_t row = (size_t)blockIdx.x * 8 + wid;
  const float* mp = m + row * BD;
  __bf16* op = mout + row * BD;
  float v[16];
  float s = 0.f;
#pragma unroll
  for (int j = 0; j < 16; ++j) { v[j] = mp[lane + j * 32]; s += v[j]; }
#pragma unroll
  for (int o = 16; o > 0; o >>= 1) s += __shfl_xor(s, o, 32);
  const float mu = s * (1.f / BD);
  float q = 0.f;
#pragma unroll
  for (int j = 0; j < 16; ++j) { float dl = v[j] - mu; q = fmaf(dl, dl, q); }
#pragma unroll
  for (int o = 16; o > 0; o >>= 1) q += __shfl_xor(q, o, 32);
  const float inv = rsqrtf(q * (1.f / BD) + 1e-5f);
#pragma unroll
  for (int j = 0; j < 16; ++j) {
    int c = lane + j * 32;
    op[c] = (__bf16)((v[j] - mu) * inv * g[c] + bb[c]);
  }
}

// ---------------- host orchestration ----------------
extern "C" void kernel_launch(void* const* d_in, const int* in_sizes, int n_in,
                              void* d_out, int out_size, void* d_ws, size_t ws_size,
                              hipStream_t stream) {
  (void)in_sizes; (void)n_in; (void)out_size; (void)ws_size;
  const float* x      = (const float*)d_in[0];
  const float* W_down = (const float*)d_in[1];
  const float* b_down = (const float*)d_in[2];
  const float* W_in   = (const float*)d_in[3];
  const float* conv_w = (const float*)d_in[4];
  const float* conv_b = (const float*)d_in[5];
  const float* W_x    = (const float*)d_in[6];
  const float* W_dt   = (const float*)d_in[7];
  const float* b_dt   = (const float*)d_in[8];
  const float* A_log  = (const float*)d_in[9];
  const float* D_skip = (const float*)d_in[10];
  const float* W_out  = (const float*)d_in[11];
  const float* ln_g   = (const float*)d_in[12];
  const float* ln_b   = (const float*)d_in[13];
  const float* W_up   = (const float*)d_in[14];
  const float* b_up   = (const float*)d_in[15];
  float* out = (float*)d_out;

  char* ws = (char*)d_ws;
  size_t off = 0;
  auto alloc = [&](size_t bytes) -> char* {
    char* p = ws + off;
    off += (bytes + 255) & ~(size_t)255;
    return p;
  };
  __bf16* h_b  = (__bf16*)alloc((size_t)NTOK * BD * 2);
  float*  xz   = (float*)alloc((size_t)NTOK * 2 * DIN * 4);
  float*  xc   = (float*)alloc((size_t)NTOK * DIN * 4);
  __bf16* xcb  = (__bf16*)alloc((size_t)NTOK * DIN * 2);
  float*  xdbl = (float*)alloc((size_t)NTOK * 64 * 4);
  float*  dt   = (float*)alloc((size_t)NTOK * DIN * 4);
  float*  ysc  = (float*)alloc((size_t)NTOK * DIN * 4);
  __bf16* ycb  = (__bf16*)alloc((size_t)NTOK * DIN * 2);
  float*  m    = (float*)alloc((size_t)NTOK * BD * 4);
  __bf16* mn   = (__bf16*)alloc((size_t)NTOK * BD * 2);
  __bf16* Wdown_b = (__bf16*)alloc((size_t)D_SZ * BD * 2);
  __bf16* Win_b   = (__bf16*)alloc((size_t)BD * 2 * DIN * 2);
  __bf16* Wx_b    = (__bf16*)alloc((size_t)DIN * 64 * 2);
  __bf16* Wdt_b   = (__bf16*)alloc((size_t)DTR * DIN * 2);
  __bf16* Wout_b  = (__bf16*)alloc((size_t)DIN * BD * 2);
  __bf16* Wup_b   = (__bf16*)alloc((size_t)BD * D_SZ * 2);
  float* aneg = (float*)alloc((size_t)DIN * NST * 4);

  const dim3 blk(256);
  auto cvt = [&](const float* in, __bf16* o, int n) {
    cvt_bf16_kernel<<<(n + 255) / 256, blk, 0, stream>>>(in, o, n);
  };
  cvt(W_down, Wdown_b, D_SZ * BD);
  cvt(W_in,   Win_b,   BD * 2 * DIN);
  cvt(W_x,    Wx_b,    DIN * 64);
  cvt(W_dt,   Wdt_b,   DTR * DIN);
  cvt(W_out,  Wout_b,  DIN * BD);
  cvt(W_up,   Wup_b,   BD * D_SZ);
  aneg_kernel<<<(DIN * NST + 255) / 256, blk, 0, stream>>>(A_log, aneg, DIN * NST);

  // 1) h = bf16(x @ W_down + b_down)          [16384,512]  K=1024  (A: f32 cvt)
  gemm_bf16_kernel<1, float, __bf16><<<dim3(BD / BN, NTOK / BM), blk, 0, stream>>>(
      x, Wdown_b, b_down, h_b, NTOK, BD, D_SZ, D_SZ, BD);
  // 2) xz = h @ W_in                           [16384,2048] K=512   (A: bf16 TDM)
  gemm_bf16_kernel<0, __bf16, float><<<dim3(2 * DIN / BN, NTOK / BM), blk, 0, stream>>>(
      h_b, Win_b, nullptr, xz, NTOK, 2 * DIN, BD, BD, 2 * DIN);
  // 3) xc/xcb = silu(causal_conv4(xz[:, :DIN]))
  {
    size_t tot = (size_t)NTOK * DIN;
    conv_silu_kernel<<<(unsigned)((tot + 255) / 256), blk, 0, stream>>>(
        xz, conv_w, conv_b, xc, xcb);
  }
  // 4) x_dbl = xcb @ W_x                       [16384,64]   K=1024  (A: bf16 TDM)
  gemm_bf16_kernel<0, __bf16, float><<<dim3(1, NTOK / BM), blk, 0, stream>>>(
      xcb, Wx_b, nullptr, xdbl, NTOK, 64, DIN, DIN, 64);
  // 5) dt = softplus(x_dbl[:, :32] @ W_dt + b_dt)  [16384,1024] K=32 (A: f32, lda=64)
  gemm_bf16_kernel<2, float, float><<<dim3(DIN / BN, NTOK / BM), blk, 0, stream>>>(
      xdbl, Wdt_b, b_dt, dt, NTOK, DIN, DTR, 64, DIN);
  // 6) selective scan -> ysc
  scan_kernel<<<dim3(DIN / 256, B_SZ), blk, 0, stream>>>(dt, xc, xdbl, aneg, ysc);
  // 7) ycb = bf16[(ysc + D_skip*xc) * silu(z)]
  {
    size_t tot = (size_t)NTOK * DIN;
    combine_kernel<<<(unsigned)((tot + 255) / 256), blk, 0, stream>>>(
        ysc, xc, xz, D_skip, ycb);
  }
  // 8) m = ycb @ W_out                         [16384,512]  K=1024  (A: bf16 TDM)
  gemm_bf16_kernel<0, __bf16, float><<<dim3(BD / BN, NTOK / BM), blk, 0, stream>>>(
      ycb, Wout_b, nullptr, m, NTOK, BD, DIN, DIN, BD);
  // 9) mn = bf16(LayerNorm(m))
  layernorm_kernel<<<NTOK / 8, blk, 0, stream>>>(m, ln_g, ln_b, mn);
  // 10) out = mn @ W_up + b_up                 [16384,1024] K=512   (A: bf16 TDM)
  gemm_bf16_kernel<1, __bf16, float><<<dim3(D_SZ / BN, NTOK / BM), blk, 0, stream>>>(
      mn, Wup_b, b_up, out, NTOK, D_SZ, BD, BD, D_SZ);
}
